// MVGGT_37323265802667
// MI455X (gfx1250) — matrix operand order repaired and checked by reading, not compiled, not run on hardware
//
#include <hip/hip_runtime.h>
#include <hip/hip_bf16.h>

typedef __attribute__((ext_vector_type(16))) _Float16 v16h;
typedef __attribute__((ext_vector_type(8)))  _Float16 v8h;
typedef __attribute__((ext_vector_type(8)))  float    v8f;
typedef __attribute__((ext_vector_type(4)))  int      v4i;

#define B_N   8
#define HW_N  5476
#define M_N   (B_N * HW_N)     // 43808
#define C_N   1024
#define LIN_N 768
#define NL_N  32
#define NH_N  8
#define DK_N  128

#if __has_builtin(__builtin_amdgcn_global_load_async_to_lds_b128)
#define PWAM_ASYNC 1
#else
#define PWAM_ASYNC 0
#endif

__device__ __forceinline__ float gelu_exact(float x) {
    return 0.5f * x * (1.0f + erff(x * 0.70710678118654752f));
}

#if PWAM_ASYNC
typedef __attribute__((address_space(1))) v4i* pwam_gp4;
typedef __attribute__((address_space(3))) v4i* pwam_lp4;
__device__ __forceinline__ void pwam_async_b128(const void* g, void* l) {
    __builtin_amdgcn_global_load_async_to_lds_b128(
        (pwam_gp4)(uintptr_t)g, (pwam_lp4)(uintptr_t)l, 0, 0);
}
__device__ __forceinline__ void pwam_wait_async() {
#if __has_builtin(__builtin_amdgcn_s_wait_asynccnt)
    __builtin_amdgcn_s_wait_asynccnt(0);
#else
    asm volatile("s_wait_asynccnt 0x0" ::: "memory");
#endif
}
#endif

// ---------------------------------------------------------------------------
// Packing kernels
// ---------------------------------------------------------------------------
__global__ void pwam_cvt_f16(const float* __restrict__ src, _Float16* __restrict__ dst, size_t n) {
    size_t stride = (size_t)gridDim.x * blockDim.x;
    for (size_t i = (size_t)blockIdx.x * blockDim.x + threadIdx.x; i < n; i += stride)
        dst[i] = (_Float16)src[i];
}

// src: [O][I] f32 row-major  ->  dst: [I][O] f16 row-major (B operand for WMMA)
__global__ void pwam_pack_wT(const float* __restrict__ src, _Float16* __restrict__ dst, int O, int I) {
    size_t n = (size_t)O * I;
    size_t stride = (size_t)gridDim.x * blockDim.x;
    for (size_t i = (size_t)blockIdx.x * blockDim.x + threadIdx.x; i < n; i += stride) {
        int o = (int)(i / I);
        int k = (int)(i % I);
        dst[(size_t)k * O + o] = (_Float16)src[i];
    }
}

// ---------------------------------------------------------------------------
// Main WMMA GEMM:  Y[M,1024] = A[M,1024] * Bt[1024,1024]  (+bias, +activation)
// MODE 0: store f16 plain    MODE 1: store f16 GELU    MODE 2: store f32 GELU
// Double-buffered LDS, one barrier per K-step, async global->LDS when available.
// ---------------------------------------------------------------------------
#define AS_STRIDE 40    // 32 + 8 halves  (80B rows, 16B aligned fragments)
#define BS_STRIDE 136   // 128 + 8 halves (272B rows, 16B aligned fragments)

template <int MODE>
__global__ __launch_bounds__(256) void pwam_gemm(const _Float16* __restrict__ A,
                                                 const _Float16* __restrict__ Bt,
                                                 const float* __restrict__ bias,
                                                 _Float16* __restrict__ out16,
                                                 float* __restrict__ out32,
                                                 int Mtot) {
    __shared__ __align__(16) _Float16 As[2][128 * AS_STRIDE];
    __shared__ __align__(16) _Float16 Bs[2][32 * BS_STRIDE];

    const int tid  = threadIdx.x;
    const int row0 = blockIdx.x * 128;
    const int n0   = blockIdx.y * 128;

    const int lane = tid & 31;
    const int w    = tid >> 5;      // wave id 0..7 -> 16-row band
    const int r    = lane & 15;
    const int sel  = lane >> 4;

    // staging roles
    const int arow = tid >> 1, aseg = tid & 1;   // A tile: 128 rows x 2 segs of 16 halves
    const int brow = tid >> 3, bseg = tid & 7;   // B tile: 32 rows x 8 segs of 16 halves
    int am = row0 + arow;
    if (am >= Mtot) am = Mtot - 1;               // clamp; padded rows masked in epilogue

    v8f acc[8] = {};

#if PWAM_ASYNC
    auto issue_tile = [&](int k0, int s) {
        const _Float16* gp = A + (size_t)am * C_N + k0 + aseg * 16;
        _Float16* lp = &As[s][arow * AS_STRIDE + aseg * 16];
        pwam_async_b128(gp, lp);
        pwam_async_b128(gp + 8, lp + 8);
        pwam_async_b128(Bt + (size_t)(k0 + brow) * C_N + n0 + bseg * 16,
                        &Bs[s][brow * BS_STRIDE + bseg * 16]);
    };
    issue_tile(0, 0);
    pwam_wait_async();
    __syncthreads();
#else
    uint4 ra0, ra1, rb0;
    auto load_regs = [&](int k0) {
        const uint4* asrc = (const uint4*)(A + (size_t)am * C_N + k0 + aseg * 16);
        ra0 = asrc[0];
        ra1 = asrc[1];
        rb0 = *(const uint4*)(Bt + (size_t)(k0 + brow) * C_N + n0 + bseg * 16);
    };
    auto store_lds = [&](int s) {
        uint4* adst = (uint4*)(&As[s][arow * AS_STRIDE + aseg * 16]);
        adst[0] = ra0;
        adst[1] = ra1;
        *(uint4*)(&Bs[s][brow * BS_STRIDE + bseg * 16]) = rb0;
    };
    load_regs(0);
    store_lds(0);
    __syncthreads();
#endif

    int cur = 0;
    for (int k0 = 0; k0 < C_N; k0 += 32) {
        const bool hasNext = (k0 + 32) < C_N;
#if PWAM_ASYNC
        if (hasNext) issue_tile(k0 + 32, cur ^ 1);   // overlaps with WMMAs below
#else
        if (hasNext) load_regs(k0 + 32);             // global loads in flight
#endif
        // ---- compute on buffer `cur` ----
        // A fragment: lane holds row M=r, K chunks [sel*8,+8) and [sel*8+16,+8)
        const _Float16* ap = &As[cur][(w * 16 + r) * AS_STRIDE + sel * 8];
        v8h a0 = *(const v8h*)ap;
        v8h a1 = *(const v8h*)(ap + 16);
        v16h af = __builtin_shufflevector(a0, a1, 0, 1, 2, 3, 4, 5, 6, 7,
                                                  8, 9, 10, 11, 12, 13, 14, 15);
        // Preload all 8 B fragments, then fire WMMAs back-to-back.
        v16h bf[8];
#pragma unroll
        for (int nt = 0; nt < 8; ++nt) {
            const _Float16* bp = &Bs[cur][nt * 16 + r];
            v16h bt;
#pragma unroll
            for (int s = 0; s < 16; ++s)
                bt[s] = bp[(sel * 8 + (s & 7) + (s >> 3) * 16) * BS_STRIDE];
            bf[nt] = bt;
        }
#pragma unroll
        for (int nt = 0; nt < 8; ++nt) {
            acc[nt] = __builtin_amdgcn_wmma_f32_16x16x32_f16(
                false, af, false, bf[nt], (short)0, acc[nt], false, false);
        }
        // ---- publish next buffer ----
#if PWAM_ASYNC
        pwam_wait_async();
#else
        if (hasNext) store_lds(cur ^ 1);
#endif
        __syncthreads();
        cur ^= 1;
    }

    // Epilogue: C/D layout -> VGPR rr: row = rr + 8*(lane>>4), col = lane&15
    const int colL    = lane & 15;
    const int rowBase = row0 + w * 16 + (lane >> 4) * 8;
#pragma unroll
    for (int nt = 0; nt < 8; ++nt) {
        int c = n0 + nt * 16 + colL;
        float bv = bias[c];
#pragma unroll
        for (int rr = 0; rr < 8; ++rr) {
            int m = rowBase + rr;
            if (m < Mtot) {
                float y = acc[nt][rr] + bv;
                size_t idx = (size_t)m * C_N + c;
                if (MODE == 2)      out32[idx] = gelu_exact(y);
                else if (MODE == 1) out16[idx] = (_Float16)gelu_exact(y);
                else                out16[idx] = (_Float16)y;
            }
        }
    }
}

// ---------------------------------------------------------------------------
// InstanceNorm stats: one block per (b, c); population variance over HW
// ---------------------------------------------------------------------------
__global__ __launch_bounds__(256) void pwam_inorm_stats(const _Float16* __restrict__ buf,
                                                        float* __restrict__ mean,
                                                        float* __restrict__ rstd) {
    int b = blockIdx.x >> 10;
    int c = blockIdx.x & 1023;
    float s = 0.f, q = 0.f;
    for (int hw = threadIdx.x; hw < HW_N; hw += 256) {
        float v = (float)buf[((size_t)(b * HW_N + hw)) * C_N + c];
        s += v;
        q += v * v;
    }
    __shared__ float ss[256], sq[256];
    ss[threadIdx.x] = s;
    sq[threadIdx.x] = q;
    __syncthreads();
    for (int st = 128; st > 0; st >>= 1) {
        if (threadIdx.x < st) {
            ss[threadIdx.x] += ss[threadIdx.x + st];
            sq[threadIdx.x] += sq[threadIdx.x + st];
        }
        __syncthreads();
    }
    if (threadIdx.x == 0) {
        float m = ss[0] * (1.0f / HW_N);
        float v = sq[0] * (1.0f / HW_N) - m * m;
        mean[blockIdx.x] = m;
        rstd[blockIdx.x] = rsqrtf(v + 1e-5f);
    }
}

// ---------------------------------------------------------------------------
// K/V projection: k = (fk_w @ l + fk_b)*mask, v likewise. Output [B,1024,32] f16
// ---------------------------------------------------------------------------
__global__ __launch_bounds__(256) void pwam_kv(const float* __restrict__ l,
                                               const float* __restrict__ lmask,
                                               const float* __restrict__ fkw,
                                               const float* __restrict__ fkb,
                                               const float* __restrict__ fvw,
                                               const float* __restrict__ fvb,
                                               _Float16* __restrict__ kbuf,
                                               _Float16* __restrict__ vbuf) {
    int b = blockIdx.x >> 5;
    int t = blockIdx.x & 31;
    __shared__ float lrow[LIN_N];
    for (int i = threadIdx.x; i < LIN_N; i += 256)
        lrow[i] = l[((size_t)(b * NL_N + t)) * LIN_N + i];
    __syncthreads();
    float mask = lmask[b * NL_N + t];
#pragma unroll
    for (int cc = 0; cc < 4; ++cc) {
        int c = threadIdx.x + cc * 256;
        const float* kw = fkw + (size_t)c * LIN_N;
        const float* vw = fvw + (size_t)c * LIN_N;
        float sk = 0.f, sv = 0.f;
        for (int j = 0; j < LIN_N; ++j) {
            float lv = lrow[j];
            sk += kw[j] * lv;
            sv += vw[j] * lv;
        }
        size_t idx = ((size_t)b * C_N + c) * NL_N + t;
        kbuf[idx] = (_Float16)((sk + fkb[c]) * mask);
        vbuf[idx] = (_Float16)((sv + fvb[c]) * mask);
    }
}

// ---------------------------------------------------------------------------
// Attention: per (b, h, 128-row tile). q normalized on load. N_l=32, dk=dv=128.
// ---------------------------------------------------------------------------
__global__ __launch_bounds__(128) void pwam_attn(const _Float16* __restrict__ qbuf,
                                                 const _Float16* __restrict__ kbuf,
                                                 const _Float16* __restrict__ vbuf,
                                                 const float* __restrict__ meanq,
                                                 const float* __restrict__ rstdq,
                                                 const float* __restrict__ lmask,
                                                 _Float16* __restrict__ obuf) {
    int b = blockIdx.z, h = blockIdx.y;
    int hw = blockIdx.x * 128 + threadIdx.x;
    __shared__ _Float16 ksm[DK_N][36];
    __shared__ _Float16 vsm[DK_N][36];
    __shared__ _Float16 qs[128][132];
    __shared__ float msk[NL_N];

    for (int i = threadIdx.x; i < DK_N * NL_N; i += 128) {
        int d = i >> 5, t = i & 31;
        size_t src = ((size_t)b * C_N + h * DK_N + d) * NL_N + t;
        ksm[d][t] = kbuf[src];
        vsm[d][t] = vbuf[src];
    }
    if (threadIdx.x < NL_N) msk[threadIdx.x] = lmask[b * NL_N + threadIdx.x];

    bool valid = hw < HW_N;
    size_t row = ((size_t)b * HW_N + (valid ? hw : 0)) * C_N + h * DK_N;
    for (int d = 0; d < DK_N; ++d) {
        float v = valid ? (float)qbuf[row + d] : 0.f;
        int c = h * DK_N + d;
        v = (v - meanq[b * C_N + c]) * rstdq[b * C_N + c];
        qs[threadIdx.x][d] = (_Float16)v;
    }
    __syncthreads();

    float sim[NL_N];
    float mx = -1e30f;
    for (int t = 0; t < NL_N; ++t) {
        float s = 0.f;
        for (int d = 0; d < DK_N; ++d)
            s += (float)qs[threadIdx.x][d] * (float)ksm[d][t];
        s = s * 0.03125f + 10000.f * msk[t] - 10000.f;  // KC^-0.5 = 1/32, masked bias
        sim[t] = s;
        mx = fmaxf(mx, s);
    }
    float sum = 0.f;
    for (int t = 0; t < NL_N; ++t) {
        sim[t] = __expf(sim[t] - mx);
        sum += sim[t];
    }
    float inv = 1.f / sum;
    for (int t = 0; t < NL_N; ++t) sim[t] *= inv;

    if (valid) {
        for (int d = 0; d < DK_N; ++d) {
            float o = 0.f;
            for (int t = 0; t < NL_N; ++t)
                o += sim[t] * (float)vsm[d][t];
            obuf[row + d] = (_Float16)o;
        }
    }
}

// ---------------------------------------------------------------------------
// Gated fusion (in place over vis): vis = vis * inorm(lang)
// ---------------------------------------------------------------------------
__global__ void pwam_fuse(_Float16* __restrict__ vis,
                          const _Float16* __restrict__ lang,
                          const float* __restrict__ meanl,
                          const float* __restrict__ rstdl,
                          size_t n) {
    size_t stride = (size_t)gridDim.x * blockDim.x;
    for (size_t i = (size_t)blockIdx.x * blockDim.x + threadIdx.x; i < n; i += stride) {
        int c = (int)(i & 1023);
        size_t rowm = i >> 10;
        int b = (int)(rowm / HW_N);
        float lg = ((float)lang[i] - meanl[b * C_N + c]) * rstdl[b * C_N + c];
        vis[i] = (_Float16)((float)vis[i] * lg);
    }
}

// ---------------------------------------------------------------------------
extern "C" void kernel_launch(void* const* d_in, const int* in_sizes, int n_in,
                              void* d_out, int out_size, void* d_ws, size_t ws_size,
                              hipStream_t stream) {
    const float* x     = (const float*)d_in[0];
    const float* l     = (const float*)d_in[1];
    const float* lmask = (const float*)d_in[2];
    const float* vis_w = (const float*)d_in[3];
    const float* vis_b = (const float*)d_in[4];
    const float* fq_w  = (const float*)d_in[5];
    const float* fq_b  = (const float*)d_in[6];
    const float* fk_w  = (const float*)d_in[7];
    const float* fk_b  = (const float*)d_in[8];
    const float* fv_w  = (const float*)d_in[9];
    const float* fv_b  = (const float*)d_in[10];
    const float* W_w   = (const float*)d_in[11];
    const float* W_b   = (const float*)d_in[12];
    const float* pm_w  = (const float*)d_in[13];
    const float* pm_b  = (const float*)d_in[14];
    float* out = (float*)d_out;

    char* wsb = (char*)d_ws;
    size_t off = 0;
    auto alloc = [&](size_t bytes) -> void* {
        off = (off + 255) & ~(size_t)255;
        void* p = wsb + off;
        off += bytes;
        return p;
    };

    const size_t big = (size_t)M_N * C_N * sizeof(_Float16);   // ~89.7 MB
    _Float16* x16   = (_Float16*)alloc(big);   // later reused as attention output
    _Float16* vis16 = (_Float16*)alloc(big);   // later overwritten in place with mm
    _Float16* q16   = (_Float16*)alloc(big);   // later reused as lang_raw
    _Float16* wtVis = (_Float16*)alloc((size_t)C_N * C_N * 2);
    _Float16* wtFq  = (_Float16*)alloc((size_t)C_N * C_N * 2);
    _Float16* wtW   = (_Float16*)alloc((size_t)C_N * C_N * 2);
    _Float16* wtPm  = (_Float16*)alloc((size_t)C_N * C_N * 2);
    _Float16* wtFk  = (_Float16*)alloc((size_t)LIN_N * C_N * 2);
    _Float16* wtFv  = (_Float16*)alloc((size_t)LIN_N * C_N * 2);
    _Float16* kbuf  = (_Float16*)alloc((size_t)B_N * C_N * NL_N * 2);
    _Float16* vbuf  = (_Float16*)alloc((size_t)B_N * C_N * NL_N * 2);
    float* meanq = (float*)alloc((size_t)B_N * C_N * 4);
    float* rstdq = (float*)alloc((size_t)B_N * C_N * 4);
    float* meanl = (float*)alloc((size_t)B_N * C_N * 4);
    float* rstdl = (float*)alloc((size_t)B_N * C_N * 4);

    _Float16* attn16 = x16;   // x no longer needed after q GEMM
    _Float16* lang16 = q16;   // q no longer needed after attention
    _Float16* mm16   = vis16; // gated fusion done in place

    // 1) pack inputs/weights to f16
    pwam_cvt_f16<<<4096, 256, 0, stream>>>(x, x16, (size_t)M_N * C_N);
    pwam_pack_wT<<<2048, 256, 0, stream>>>(vis_w, wtVis, C_N, C_N);
    pwam_pack_wT<<<2048, 256, 0, stream>>>(fq_w,  wtFq,  C_N, C_N);
    pwam_pack_wT<<<2048, 256, 0, stream>>>(W_w,   wtW,   C_N, C_N);
    pwam_pack_wT<<<2048, 256, 0, stream>>>(pm_w,  wtPm,  C_N, C_N);
    pwam_pack_wT<<<2048, 256, 0, stream>>>(fk_w,  wtFk,  C_N, LIN_N);
    pwam_pack_wT<<<2048, 256, 0, stream>>>(fv_w,  wtFv,  C_N, LIN_N);

    dim3 ggrid((M_N + 127) / 128, C_N / 128);

    // 2) vis = GELU(x @ vis_w^T + b),  q_raw = x @ fq_w^T + b
    pwam_gemm<1><<<ggrid, 256, 0, stream>>>(x16, wtVis, vis_b, vis16, nullptr, M_N);
    pwam_gemm<0><<<ggrid, 256, 0, stream>>>(x16, wtFq,  fq_b,  q16,   nullptr, M_N);

    // 3) InstanceNorm stats for q
    pwam_inorm_stats<<<B_N * C_N, 256, 0, stream>>>(q16, meanq, rstdq);

    // 4) masked language K/V
    pwam_kv<<<B_N * NL_N, 256, 0, stream>>>(l, lmask, fk_w, fk_b, fv_w, fv_b, kbuf, vbuf);

    // 5) attention (q normalized on load) -> attn16 (reuses x16)
    dim3 agrid((HW_N + 127) / 128, NH_N, B_N);
    pwam_attn<<<agrid, 128, 0, stream>>>(q16, kbuf, vbuf, meanq, rstdq, lmask, attn16);

    // 6) lang_raw = attn @ W_w^T + b  (reuses q16)
    pwam_gemm<0><<<ggrid, 256, 0, stream>>>(attn16, wtW, W_b, lang16, nullptr, M_N);

    // 7) InstanceNorm stats for lang
    pwam_inorm_stats<<<B_N * C_N, 256, 0, stream>>>(lang16, meanl, rstdl);

    // 8) mm = vis * inorm(lang)  (in place over vis16)
    pwam_fuse<<<4096, 256, 0, stream>>>(vis16, lang16, meanl, rstdl, (size_t)M_N * C_N);

    // 9) out = GELU(mm @ pm_w^T + b)  -> fp32 [B, HW, VC]
    pwam_gemm<2><<<ggrid, 256, 0, stream>>>(mm16, wtPm, pm_b, nullptr, out, M_N);

    (void)in_sizes; (void)n_in; (void)out_size; (void)ws_size;
}
